// HemisphericPairExpert_34359738586
// MI455X (gfx1250) — compile-verified, matrix operand order checked by reading, not run on hardware
//
#include <hip/hip_runtime.h>
#include <hip/hip_bf16.h>
#include <math.h>

// Problem dims (B,T,C,D)=(16,512,64,128), H=256
#define DIM_B 16
#define DIM_T 512
#define DIM_C 64
#define DIM_D 128
#define DIM_H 256
#define NSLICE (DIM_B * DIM_T)
#define NTHREADS 256
#define NWAVES 8
#define LN_EPS 1e-5f

typedef __bf16 bf16;
typedef __attribute__((ext_vector_type(16))) __bf16 v16bf;
typedef __attribute__((ext_vector_type(8)))  float  v8f;

// ---------------------------------------------------------------------------
// WMMA fragment layout helpers (CDNA5 ISA 7.12.2, 16-bit A 16x32 / B 32x16):
// lane half (0:lanes 0-15, 1:lanes 16-31) holds K-groups {0..7,16..23} /
// {8..15,24..31}; 16 bf16 elems per lane -> one contiguous 32B chunk when
// pre-swizzled, so a fragment load is two ds_load_b128.
// ---------------------------------------------------------------------------
__device__ __forceinline__ int staged_idx(int mn, int k, int ksteps) {
  int kk   = k & 31;
  int half = (kk >> 3) & 1;
  int i    = (kk & 7) + ((kk & 16) ? 8 : 0);
  return ((((mn >> 4) * ksteps + (k >> 5)) << 5) + (half << 4) + (mn & 15)) * 16 + i;
}

__device__ __forceinline__ v16bf load_frag(const bf16* base, int tile, int kstep,
                                           int ksteps, int lane) {
  return *(const v16bf*)(base + (((tile * ksteps + kstep) << 5) + lane) * 16);
}

__device__ __forceinline__ v8f wmma_bf16f32(v16bf a, v16bf b, v8f c) {
  // (neg_a, A, neg_b, B, c_mod, C, reuse_a, reuse_b)
  return __builtin_amdgcn_wmma_f32_16x16x32_bf16(false, a, false, b, (short)0, c,
                                                 false, false);
}

// Async global -> LDS copy, 16B per lane (gfx1250 VFLAT async op, ASYNCcnt).
__device__ __forceinline__ void async_copy_b128(void* lds_dst, const void* gaddr) {
  // LDS aperture: LDS byte address == low 32 bits of the flat pointer.
  unsigned ldsOff = (unsigned)(size_t)lds_dst;
  asm volatile("global_load_async_to_lds_b128 %0, %1, off"
               :: "v"(ldsOff), "v"((unsigned long long)(size_t)gaddr)
               : "memory");
}

__device__ __forceinline__ float wave_sum32(float v) {
#pragma unroll
  for (int m = 16; m >= 1; m >>= 1) v += __shfl_xor(v, m, 32);
  return v;
}

// Branch-free exact-gelu: Abramowitz-Stegun 7.1.26 erf (|err| < 1.5e-7),
// v_rcp_f32 + v_exp_f32 + fma chain; no EXEC divergence inside WMMA loops.
__device__ __forceinline__ float gelu_exact(float x) {
  float z  = x * 0.70710678118654752440f;
  float az = __builtin_fabsf(z);
  float t  = __builtin_amdgcn_rcpf(__builtin_fmaf(0.3275911f, az, 1.0f));
  float p  = __builtin_fmaf(1.061405429f, t, -1.453152027f);
  p        = __builtin_fmaf(p, t, 1.421413741f);
  p        = __builtin_fmaf(p, t, -0.284496736f);
  p        = __builtin_fmaf(p, t, 0.254829592f);
  p        = p * t;
  float e  = __expf(-az * az);
  float er = __builtin_copysignf(__builtin_fmaf(-p, e, 1.0f), z);
  return 0.5f * x * (1.0f + er);
}

// Stage a KxN fp32 row-major weight matrix from global (L2-resident) into LDS
// as bf16 in WMMA B-fragment order. Coalesced float4 global reads, scattered
// 2-byte LDS writes (conflict-free: consecutive lanes of a 16-block).
__device__ __forceinline__ void stage_weight(const float* __restrict__ Wg, bf16* dst,
                                             int K, int N, int logN, int tid) {
  const int ksteps = K >> 5;
  const int total  = K * N;
  for (int idx = tid * 4; idx < total; idx += NTHREADS * 4) {
    const float4 v = *(const float4*)(Wg + idx);
    int n = idx & (N - 1);
    int k = idx >> logN;
    int kk = k & 31;
    int half = (kk >> 3) & 1;
    int i = (kk & 7) + ((kk & 16) ? 8 : 0);
    int base = ((((n >> 4) * ksteps + (k >> 5)) << 5) + (half << 4) + (n & 15)) * 16 + i;
    dst[base +  0] = (bf16)v.x;
    dst[base + 16] = (bf16)v.y;
    dst[base + 32] = (bf16)v.z;
    dst[base + 48] = (bf16)v.w;
  }
}

// One GEMM pass: M=64 (mtile = wave&3 fixed per wave), N = NTILES*16,
// K = KSTEPS*32. sBc used for mtiles 0-1, sBd for mtiles 2-3 (same ptr for
// single-weight GEMMs). Bias preloaded to registers, accumulator-initialized.
// Output: GELU->bf16 fragments (feeds next GEMM) or raw f32 row-major.
// n-tile loop fully unrolled so ds_load/wmma chains software-pipeline.
template <int NTILES, int KSTEPS, bool GELU_STAGED, int OUTKS>
__device__ __forceinline__ void run_gemm(const bf16* sA, const bf16* sBc,
                                         const bf16* sBd,
                                         const float* __restrict__ biasC,
                                         const float* __restrict__ biasD,
                                         bf16* outStaged, float* outF32,
                                         int wave, int lane) {
  constexpr int JT = NTILES / 2;  // tiles per wave
  const int   mtile = wave & 3;
  const int   n0    = wave >> 2;
  const bf16* sB    = (mtile < 2) ? sBc : sBd;
  const float* bias = (mtile < 2) ? biasC : biasD;

  // Preload biases (global loads pipeline together, no in-loop waits).
  float bv[JT];
#pragma unroll
  for (int j = 0; j < JT; ++j) bv[j] = bias[(n0 + 2 * j) * 16 + (lane & 15)];

  // A fragments for this wave's fixed mtile, reused across all n-tiles.
  v16bf afrag[KSTEPS];
#pragma unroll
  for (int ks = 0; ks < KSTEPS; ++ks) afrag[ks] = load_frag(sA, mtile, ks, KSTEPS, lane);

#pragma unroll
  for (int j = 0; j < JT; ++j) {
    const int ntile = n0 + 2 * j;
    v8f acc;
#pragma unroll
    for (int r = 0; r < 8; ++r) acc[r] = bv[j];
#pragma unroll
    for (int ks = 0; ks < KSTEPS; ++ks) {
      v16bf bfrag = load_frag(sB, ntile, ks, KSTEPS, lane);
      acc = wmma_bf16f32(afrag[ks], bfrag, acc);
    }
    // C/D layout: vgpr r -> M = mtile*16 + (lane>=16 ? 8 : 0) + r, N = lane&15
    const int ncol  = ntile * 16 + (lane & 15);
    const int mbase = mtile * 16 + ((lane >> 4) << 3);
    if constexpr (GELU_STAGED) {
#pragma unroll
      for (int r = 0; r < 8; ++r)
        outStaged[staged_idx(mbase + r, ncol, OUTKS)] = (bf16)gelu_exact(acc[r]);
    } else {
#pragma unroll
      for (int r = 0; r < 8; ++r)
        outF32[(mbase + r) * (NTILES * 16) + ncol] = acc[r];
    }
  }
}

__global__ __launch_bounds__(NTHREADS, 1) void hemi_pair_expert_kernel(
    const float* __restrict__ x,
    const float* __restrict__ ln_g, const float* __restrict__ ln_b,
    const float* __restrict__ cW1, const float* __restrict__ cb1,
    const float* __restrict__ cW2, const float* __restrict__ cb2,
    const float* __restrict__ dW1, const float* __restrict__ db1,
    const float* __restrict__ dW2, const float* __restrict__ db2,
    const float* __restrict__ fW1, const float* __restrict__ fb1,
    const float* __restrict__ fW2, const float* __restrict__ fb2,
    float* __restrict__ out) {
  __shared__ __align__(16) float s_x[DIM_C * DIM_D];     // 32 KB: x slice -> out slice
  __shared__ __align__(32) bf16  s_w[2 * DIM_D * DIM_D]; // 64 KB: staged weights (per phase)
  __shared__ __align__(32) bf16  s_a[DIM_C * DIM_D];     // 16 KB: A fragments
  __shared__ __align__(32) bf16  s_h[DIM_C * DIM_H];     // 32 KB: hidden fragments
  __shared__ __align__(16) float s_u[DIM_C * DIM_D];     // 32 KB: GEMM f32 outputs

  const int tid   = threadIdx.x;
  const int lane  = tid & 31;
  const int wave  = tid >> 5;
  const int slice = blockIdx.x;

  const float* xg   = x + (size_t)slice * (DIM_C * DIM_D);
  float*       outg = out + (size_t)slice * (DIM_C * DIM_D);

  // Hint the L2-resident weights into cache (global_prefetch_b8).
  if (tid == 0) {
    __builtin_prefetch(cW1, 0, 1);
    __builtin_prefetch(dW1, 0, 1);
    __builtin_prefetch(cW2, 0, 1);
    __builtin_prefetch(dW2, 0, 1);
    __builtin_prefetch(fW1, 0, 1);
    __builtin_prefetch(fW2, 0, 1);
  }

  // ---- Phase 0: async-DMA x slice to LDS while staging cW1/dW1 (VMEM) ----
  for (int idx = tid * 4; idx < DIM_C * DIM_D; idx += NTHREADS * 4)
    async_copy_b128(s_x + idx, xg + idx);
  stage_weight(cW1, s_w,               DIM_D, DIM_D, 7, tid);
  stage_weight(dW1, s_w + DIM_D*DIM_D, DIM_D, DIM_D, 7, tid);
  asm volatile("s_wait_asynccnt 0x0" ::: "memory");
  __syncthreads();

  // ---- Phase 1: LN row pairs -> common (rows 0-31) / diff (rows 32-63) ----
  {
    const float4 gv = *(const float4*)(ln_g + lane * 4);
    const float4 bv = *(const float4*)(ln_b + lane * 4);
    const float gj[4] = {gv.x, gv.y, gv.z, gv.w};
    const float bj[4] = {bv.x, bv.y, bv.z, bv.w};
#pragma unroll
    for (int q = 0; q < 4; ++q) {
      const int p = wave * 4 + q;  // pair index 0..31
      const float4 x0 = *(const float4*)(s_x + (2 * p) * DIM_D + lane * 4);
      const float4 x1 = *(const float4*)(s_x + (2 * p) * DIM_D + DIM_D + lane * 4);
      float a0[4] = {x0.x, x0.y, x0.z, x0.w};
      float a1[4] = {x1.x, x1.y, x1.z, x1.w};
      float mu0 = wave_sum32(a0[0] + a0[1] + a0[2] + a0[3]) * (1.0f / DIM_D);
      float mu1 = wave_sum32(a1[0] + a1[1] + a1[2] + a1[3]) * (1.0f / DIM_D);
      float v0 = 0.f, v1 = 0.f;
#pragma unroll
      for (int j = 0; j < 4; ++j) {
        a0[j] -= mu0; v0 += a0[j] * a0[j];
        a1[j] -= mu1; v1 += a1[j] * a1[j];
      }
      float rs0 = rsqrtf(wave_sum32(v0) * (1.0f / DIM_D) + LN_EPS);
      float rs1 = rsqrtf(wave_sum32(v1) * (1.0f / DIM_D) + LN_EPS);
#pragma unroll
      for (int j = 0; j < 4; ++j) {
        float h0 = a0[j] * rs0 * gj[j] + bj[j];
        float h1 = a1[j] * rs1 * gj[j] + bj[j];
        int d = lane * 4 + j;
        s_a[staged_idx(p,      d, 4)] = (bf16)(0.5f * (h0 + h1));  // common
        s_a[staged_idx(32 + p, d, 4)] = (bf16)(0.5f * (h0 - h1));  // diff
      }
    }
  }
  __syncthreads();

  // ---- Phase 2: GEMM A1 (c/d first layers) + GELU -> hidden fragments ----
  run_gemm<8, 4, true, 4>(s_a, s_w, s_w + DIM_D * DIM_D, cb1, db1,
                          s_h, nullptr, wave, lane);
  __syncthreads();

  // ---- Phase 3: restage cW2/dW2, then GEMM A2 -> s_u (f32) ----
  stage_weight(cW2, s_w,               DIM_D, DIM_D, 7, tid);
  stage_weight(dW2, s_w + DIM_D*DIM_D, DIM_D, DIM_D, 7, tid);
  __syncthreads();
  run_gemm<8, 4, false, 4>(s_h, s_w, s_w + DIM_D * DIM_D, cb2, db2,
                           nullptr, s_u, wave, lane);
  __syncthreads();

  // ---- Phase 4: recombine out = x + (common_u +/- diff_u); stage fW1 ----
  for (int idx = tid; idx < 32 * DIM_D; idx += NTHREADS) {
    int c = idx >> 7, d = idx & (DIM_D - 1);
    float cu = s_u[c * DIM_D + d];
    float du = s_u[(32 + c) * DIM_D + d];
    s_x[(2 * c) * DIM_D + d]         += cu + du;
    s_x[(2 * c) * DIM_D + DIM_D + d] += cu - du;
  }
  stage_weight(fW1, s_w, DIM_D, DIM_H, 8, tid);  // 128x256 -> 64 KB
  __syncthreads();

  // ---- Phase 5: LN(out) -> A fragments ----
  {
    const float4 gv = *(const float4*)(ln_g + lane * 4);
    const float4 bv = *(const float4*)(ln_b + lane * 4);
    const float gj[4] = {gv.x, gv.y, gv.z, gv.w};
    const float bj[4] = {bv.x, bv.y, bv.z, bv.w};
#pragma unroll
    for (int q = 0; q < 8; ++q) {
      const int m = wave * 8 + q;
      const float4 xv = *(const float4*)(s_x + m * DIM_D + lane * 4);
      float a[4] = {xv.x, xv.y, xv.z, xv.w};
      float mu = wave_sum32(a[0] + a[1] + a[2] + a[3]) * (1.0f / DIM_D);
      float v = 0.f;
#pragma unroll
      for (int j = 0; j < 4; ++j) { a[j] -= mu; v += a[j] * a[j]; }
      float rs = rsqrtf(wave_sum32(v) * (1.0f / DIM_D) + LN_EPS);
#pragma unroll
      for (int j = 0; j < 4; ++j)
        s_a[staged_idx(m, lane * 4 + j, 4)] = (bf16)(a[j] * rs * gj[j] + bj[j]);
    }
  }
  __syncthreads();

  // ---- Phase 6: GEMM F1 (128->256) + GELU -> hidden fragments (K=256) ----
  run_gemm<16, 4, true, 8>(s_a, s_w, s_w, fb1, fb1, s_h, nullptr, wave, lane);
  __syncthreads();

  // ---- Phase 7: stage fW2, GEMM F2 (256->128) -> s_u ----
  stage_weight(fW2, s_w, DIM_H, DIM_D, 7, tid);  // 256x128 -> 64 KB
  __syncthreads();
  run_gemm<8, 8, false, 0>(s_h, s_w, s_w, fb2, fb2, nullptr, s_u, wave, lane);
  __syncthreads();

  // ---- Phase 8: residual add + store (coalesced b128) ----
  for (int idx = tid * 4; idx < DIM_C * DIM_D; idx += NTHREADS * 4) {
    float4 xv = *(const float4*)(s_x + idx);
    float4 uv = *(const float4*)(s_u + idx);
    float4 o;
    o.x = xv.x + uv.x; o.y = xv.y + uv.y; o.z = xv.z + uv.z; o.w = xv.w + uv.w;
    *(float4*)(outg + idx) = o;
  }
}

extern "C" void kernel_launch(void* const* d_in, const int* in_sizes, int n_in,
                              void* d_out, int out_size, void* d_ws, size_t ws_size,
                              hipStream_t stream) {
  (void)in_sizes; (void)n_in; (void)out_size; (void)d_ws; (void)ws_size;
  const float* x    = (const float*)d_in[0];
  const float* ln_g = (const float*)d_in[1];
  const float* ln_b = (const float*)d_in[2];
  const float* cW1  = (const float*)d_in[3];
  const float* cb1  = (const float*)d_in[4];
  const float* cW2  = (const float*)d_in[5];
  const float* cb2  = (const float*)d_in[6];
  const float* dW1  = (const float*)d_in[7];
  const float* db1  = (const float*)d_in[8];
  const float* dW2  = (const float*)d_in[9];
  const float* db2  = (const float*)d_in[10];
  const float* fW1  = (const float*)d_in[11];
  const float* fb1  = (const float*)d_in[12];
  const float* fW2  = (const float*)d_in[13];
  const float* fb2  = (const float*)d_in[14];
  // left_idx / right_idx (d_in[15], d_in[16]) are the fixed even/odd pairing.

  hipLaunchKernelGGL(hemi_pair_expert_kernel, dim3(NSLICE), dim3(NTHREADS), 0, stream,
                     x, ln_g, ln_b, cW1, cb1, cW2, cb2, dW1, db1, dW2, db2,
                     fW1, fb1, fW2, fb2, (float*)d_out);
}
